// CrossAttention_12678743458362
// MI455X (gfx1250) — compile-verified, matrix operand order checked by reading, not run on hardware
//
#include <hip/hip_runtime.h>
#include <hip/hip_bf16.h>

typedef __attribute__((ext_vector_type(16))) _Float16 v16h;
typedef __attribute__((ext_vector_type(8)))  _Float16 v8h;
typedef __attribute__((ext_vector_type(8)))  float    v8f;
typedef __attribute__((ext_vector_type(4)))  int      v4i;

#define D_EMB    1024
#define N_HEADS  16
#define HEAD_DIM 64
#define T_SEQ    1024
#define BATCH    4
#define M_TOT    (BATCH * T_SEQ)   // 4096

#if defined(__has_builtin)
#if __has_builtin(__builtin_amdgcn_global_load_async_to_lds_b128)
#define HAVE_ASYNC_LDS 1
#endif
#endif

#if HAVE_ASYNC_LDS
typedef __attribute__((address_space(1))) v4i* gvec_ptr;
typedef __attribute__((address_space(3))) v4i* lvec_ptr;
#endif

// ---------------------------------------------------------------------------
// WMMA helpers (gfx1250, wave32): D = A(16x32 f16) * B(32x16 f16) + C(f32)
// ---------------------------------------------------------------------------
static __device__ __forceinline__ v8f wmma_f16(v16h a, v16h b, v8f c) {
  return __builtin_amdgcn_wmma_f32_16x16x32_f16(
      /*neg_a=*/false, a, /*neg_b=*/false, b,
      /*c_mod=*/(short)0, c, /*reuse_a=*/false, /*reuse_b=*/false);
}

static __device__ __forceinline__ v16h combine8(v8h lo, v8h hi) {
  v16h r;
#pragma unroll
  for (int i = 0; i < 8; ++i) { r[i] = lo[i]; r[i + 8] = hi[i]; }
  return r;
}

// A-fragment lane mapping (16-bit A 16x32): lane l holds row m=l&15,
// K = kbase+0..7 (elems 0..7) and kbase+16..kbase+23 (elems 8..15),
// kbase = 8*(l>=16).  B-fragment is the mirrored layout (col n=l&15, same K
// mapping), so a row-major [N][K] matrix loads directly as B.
// Both chunks are contiguous 16-byte loads (global b128 or ds b128).
static __device__ __forceinline__ v16h
load_frag(const _Float16* __restrict__ base, int stride, int row, int kbase) {
  const _Float16* p = base + (size_t)row * stride + kbase;
  v8h lo = *(const v8h*)(p);
  v8h hi = *(const v8h*)(p + 16);
  return combine8(lo, hi);
}

static __device__ __forceinline__ void wait_async0() {
#if defined(__has_builtin) && __has_builtin(__builtin_amdgcn_s_wait_asynccnt)
  __builtin_amdgcn_s_wait_asynccnt(0);
#else
  asm volatile("s_wait_asynccnt 0" ::: "memory");
#endif
}

// ---------------------------------------------------------------------------
// fp32 -> f16 elementwise cast
// ---------------------------------------------------------------------------
__global__ void __launch_bounds__(256)
cast_f16_kernel(const float* __restrict__ src, _Float16* __restrict__ dst, int n) {
  int i = blockIdx.x * blockDim.x + threadIdx.x;
  int stride = gridDim.x * blockDim.x;
  for (; i < n; i += stride) dst[i] = (_Float16)src[i];
}

// ---------------------------------------------------------------------------
// GEMM: C[M=4096, N=1024] = A[M,K=1024] (f16 rm) x W[N,K] (f16 rm) + bias[N]
// Block: 256 thr = 8 waves; wave tile 32(M)x64(N); block tile 128x128.
// W tile (128 rows x 32 k = 8KB) is double-buffer staged into LDS with
// GLOBAL_LOAD_ASYNC_TO_LDS_B128 (ASYNCcnt) and consumed via ds_load_b128;
// falls back to direct global B-fragment loads if the builtin is absent.
// MODE 0: q16 [B,H,T,Dh] | 1: k16 [B,H,T,Dh] | 2: vT16 [B,H,Dh,T] | 3: f32 out
// ---------------------------------------------------------------------------
#if HAVE_ASYNC_LDS
static __device__ __forceinline__ void
stage_w_tile(const _Float16* __restrict__ Wsrc, _Float16* lds_buf, int tid) {
#pragma unroll
  for (int s = 0; s < 2; ++s) {
    const int q   = tid * 2 + s;     // 512 chunks of 8 halfs = 128 rows x 32 k
    const int row = q >> 2;
    const int off = (q & 3) * 8;
    const _Float16* g = Wsrc + (size_t)row * D_EMB + off;
    _Float16*       l = lds_buf + row * 32 + off;
    __builtin_amdgcn_global_load_async_to_lds_b128(
        (gvec_ptr)g, (lvec_ptr)l, /*offset=*/0, /*cpol=*/0);
  }
}
#endif

template <int MODE>
__global__ void __launch_bounds__(256)
gemm_kernel(const _Float16* __restrict__ A, const _Float16* __restrict__ W,
            const float* __restrict__ bias, void* __restrict__ outp) {
#if HAVE_ASYNC_LDS
  __shared__ __align__(16) _Float16 wtile[2][128 * 32];   // 16 KB
#endif
  const int tid   = threadIdx.x;
  const int lane  = tid & 31;
  const int wave  = tid >> 5;
  const int half  = lane >> 4;
  const int l15   = lane & 15;
  const int kbase = half * 8;

  const int m0    = blockIdx.x * 128 + (wave & 3) * 32;
  const int n0blk = blockIdx.y * 128;
  const int n0    = n0blk + (wave >> 2) * 64;

  v8f acc[2][4] = {};

#if HAVE_ASYNC_LDS
  stage_w_tile(W + (size_t)n0blk * D_EMB, wtile[0], tid);
#endif

  for (int it = 0; it < D_EMB / 32; ++it) {
    const int kk = it * 32;
#if HAVE_ASYNC_LDS
    wait_async0();
    __syncthreads();
    if (it + 1 < D_EMB / 32)
      stage_w_tile(W + (size_t)n0blk * D_EMB + kk + 32, wtile[(it + 1) & 1], tid);
    const _Float16* Wt = wtile[it & 1];
    const int wstride = 32;
    const int nb = (wave >> 2) * 64;     // row base inside staged tile
#else
    const _Float16* Wt = W + kk;
    const int wstride = D_EMB;
    const int nb = n0;
#endif
    const v16h a0 = load_frag(A + kk, D_EMB, m0 + l15, kbase);
    const v16h a1 = load_frag(A + kk, D_EMB, m0 + 16 + l15, kbase);
#pragma unroll
    for (int j = 0; j < 4; ++j) {
      const v16h b = load_frag(Wt, wstride, nb + 16 * j + l15, kbase);
      acc[0][j] = wmma_f16(a0, b, acc[0][j]);
      acc[1][j] = wmma_f16(a1, b, acc[1][j]);
    }
  }

  // C/D layout: vgpr r, lanes 0-15 -> (M=r, N=lane); lanes 16-31 -> (M=r+8)
#pragma unroll
  for (int i = 0; i < 2; ++i) {
#pragma unroll
    for (int j = 0; j < 4; ++j) {
      const int n   = n0 + 16 * j + l15;
      const float bn = bias[n];
#pragma unroll
      for (int r = 0; r < 8; ++r) {
        const int m    = m0 + 16 * i + r + 8 * half;
        const float val = acc[i][j][r] + bn;
        if (MODE == 3) {
          ((float*)outp)[(size_t)m * D_EMB + n] = val;
        } else {
          const int bb = m >> 10, t = m & 1023;
          const int h  = n >> 6,  d = n & 63;
          if (MODE == 2) {
            ((_Float16*)outp)[((size_t)(bb * N_HEADS + h) * HEAD_DIM + d) * T_SEQ + t] =
                (_Float16)val;
          } else {
            ((_Float16*)outp)[((size_t)(bb * N_HEADS + h) * T_SEQ + t) * HEAD_DIM + d] =
                (_Float16)val;
          }
        }
      }
    }
  }
}

// ---------------------------------------------------------------------------
// Flash attention (causal), one (b,h) & 64 q-rows per block; 4 waves,
// 16 q-rows per wave; kv tiles of 32; online softmax in fp32.
// ---------------------------------------------------------------------------
__global__ void __launch_bounds__(128)
attn_kernel(const _Float16* __restrict__ q16, const _Float16* __restrict__ k16,
            const _Float16* __restrict__ vT16, _Float16* __restrict__ attn16) {
  __shared__ __align__(16) _Float16 lds[4][16 * 32];

  const int lane  = threadIdx.x & 31;
  const int wave  = threadIdx.x >> 5;
  const int half  = lane >> 4;
  const int l15   = lane & 15;
  const int kbase = half * 8;

  const int bh   = blockIdx.x >> 4;   // b*N_HEADS + h
  const int qblk = blockIdx.x & 15;
  const int qw0  = qblk * 64 + wave * 16;

  const _Float16* Q  = q16  + (size_t)bh * T_SEQ * HEAD_DIM;
  const _Float16* Kc = k16  + (size_t)bh * T_SEQ * HEAD_DIM;
  const _Float16* Vt = vT16 + (size_t)bh * HEAD_DIM * T_SEQ;
  _Float16* myLds = lds[wave];

  // Q A-fragments are invariant over the kv loop (K-dim = head_dim = 64)
  const v16h aq0 = load_frag(Q,      HEAD_DIM, qw0 + l15, kbase);
  const v16h aq1 = load_frag(Q + 32, HEAD_DIM, qw0 + l15, kbase);

  v8f acc0 = {}, acc1 = {}, acc2 = {}, acc3 = {};
  float rmax[8], rsum[8];
#pragma unroll
  for (int r = 0; r < 8; ++r) { rmax[r] = -3.0e38f; rsum[r] = 0.0f; }

  for (int k0 = 0; k0 <= qw0 + 15; k0 += 32) {
    // S[16 x 32] = Q(16x64) * K^T(64x32): K-cache rows are B-fragments direct.
    v8f s0 = {}, s1 = {};
    {
      const v16h b0 = load_frag(Kc,      HEAD_DIM, k0 + l15, kbase);
      const v16h b1 = load_frag(Kc + 32, HEAD_DIM, k0 + l15, kbase);
      s0 = wmma_f16(aq0, b0, s0);
      s0 = wmma_f16(aq1, b1, s0);
    }
    {
      const v16h b0 = load_frag(Kc,      HEAD_DIM, k0 + 16 + l15, kbase);
      const v16h b1 = load_frag(Kc + 32, HEAD_DIM, k0 + 16 + l15, kbase);
      s1 = wmma_f16(aq0, b0, s1);
      s1 = wmma_f16(aq1, b1, s1);
    }

    const bool needMask = (k0 + 31 > qw0);   // wave-uniform: diagonal tiles only
    const int n0g = k0 + l15;
    const int n1g = k0 + 16 + l15;
#pragma unroll
    for (int r = 0; r < 8; ++r) {
      float e0 = s0[r] * 0.125f;             // 1/sqrt(64)
      float e1 = s1[r] * 0.125f;
      if (needMask) {
        const int mg = qw0 + r + 8 * half;   // global q row
        if (n0g > mg) e0 = -3.0e38f;
        if (n1g > mg) e1 = -3.0e38f;
      }
      float tm = fmaxf(e0, e1);
#pragma unroll
      for (int off = 1; off < 16; off <<= 1)  // row spans 16 lanes
        tm = fmaxf(tm, __shfl_xor(tm, off, 32));
      const float nm    = fmaxf(rmax[r], tm);
      const float alpha = __expf(rmax[r] - nm);
      rmax[r] = nm;
      const float p0 = __expf(e0 - nm);
      const float p1 = __expf(e1 - nm);
      float ps = p0 + p1;
#pragma unroll
      for (int off = 1; off < 16; off <<= 1)
        ps += __shfl_xor(ps, off, 32);
      rsum[r] = rsum[r] * alpha + ps;
      acc0[r] *= alpha; acc1[r] *= alpha; acc2[r] *= alpha; acc3[r] *= alpha;
      // stage P (D-layout) into per-wave LDS tile [16 rows x 32 kv]
      myLds[(r + 8 * half) * 32 + l15]      = (_Float16)p0;
      myLds[(r + 8 * half) * 32 + 16 + l15] = (_Float16)p1;
    }
    asm volatile("s_wait_dscnt 0" ::: "memory");   // in-wave LDS transpose sync

    // PV: A = P(16x32) from LDS; B = Vt rows (contiguous in t) direct.
    const v16h ap = load_frag(myLds, 32, l15, kbase);
    acc0 = wmma_f16(ap, load_frag(Vt + k0, T_SEQ,  0 + l15, kbase), acc0);
    acc1 = wmma_f16(ap, load_frag(Vt + k0, T_SEQ, 16 + l15, kbase), acc1);
    acc2 = wmma_f16(ap, load_frag(Vt + k0, T_SEQ, 32 + l15, kbase), acc2);
    acc3 = wmma_f16(ap, load_frag(Vt + k0, T_SEQ, 48 + l15, kbase), acc3);
  }

  const int b = bh >> 4, h = bh & 15;
#pragma unroll
  for (int r = 0; r < 8; ++r) {
    const int t = qw0 + r + 8 * half;
    const float inv = 1.0f / rsum[r];
    const size_t o = ((size_t)(b * T_SEQ + t)) * D_EMB + h * HEAD_DIM + l15;
    attn16[o +  0] = (_Float16)(acc0[r] * inv);
    attn16[o + 16] = (_Float16)(acc1[r] * inv);
    attn16[o + 32] = (_Float16)(acc2[r] * inv);
    attn16[o + 48] = (_Float16)(acc3[r] * inv);
  }
}

// ---------------------------------------------------------------------------
extern "C" void kernel_launch(void* const* d_in, const int* in_sizes, int n_in,
                              void* d_out, int out_size, void* d_ws, size_t ws_size,
                              hipStream_t stream) {
  (void)in_sizes; (void)n_in; (void)out_size; (void)ws_size;
  const float* x_enc = (const float*)d_in[0];
  const float* x_dec = (const float*)d_in[1];
  const float* Wq = (const float*)d_in[2];
  const float* bq = (const float*)d_in[3];
  const float* Wk = (const float*)d_in[4];
  const float* bk = (const float*)d_in[5];
  const float* Wv = (const float*)d_in[6];
  const float* bv = (const float*)d_in[7];
  const float* Wp = (const float*)d_in[8];
  const float* bp = (const float*)d_in[9];

  const size_t NX = (size_t)M_TOT * D_EMB;  // 4M halfs
  const size_t NW = (size_t)D_EMB * D_EMB;  // 1M halfs
  _Float16* p    = (_Float16*)d_ws;
  _Float16* xd16 = p; p += NX;
  _Float16* xe16 = p; p += NX;
  _Float16* wq16 = p; p += NW;
  _Float16* wk16 = p; p += NW;
  _Float16* wv16 = p; p += NW;
  _Float16* wp16 = p; p += NW;
  _Float16* q16  = p; p += NX;
  _Float16* k16  = p; p += NX;
  _Float16* vT16 = p; p += NX;
  _Float16* at16 = p; p += NX;

  cast_f16_kernel<<<2048, 256, 0, stream>>>(x_dec, xd16, (int)NX);
  cast_f16_kernel<<<2048, 256, 0, stream>>>(x_enc, xe16, (int)NX);
  cast_f16_kernel<<<1024, 256, 0, stream>>>(Wq, wq16, (int)NW);
  cast_f16_kernel<<<1024, 256, 0, stream>>>(Wk, wk16, (int)NW);
  cast_f16_kernel<<<1024, 256, 0, stream>>>(Wv, wv16, (int)NW);
  cast_f16_kernel<<<1024, 256, 0, stream>>>(Wp, wp16, (int)NW);

  const dim3 gGemm(M_TOT / 128, D_EMB / 128);
  gemm_kernel<0><<<gGemm, 256, 0, stream>>>(xd16, wq16, bq, q16);
  gemm_kernel<1><<<gGemm, 256, 0, stream>>>(xe16, wk16, bk, k16);
  gemm_kernel<2><<<gGemm, 256, 0, stream>>>(xe16, wv16, bv, vT16);

  attn_kernel<<<BATCH * N_HEADS * (T_SEQ / 64), 128, 0, stream>>>(q16, k16, vT16, at16);

  gemm_kernel<3><<<gGemm, 256, 0, stream>>>(at16, wp16, bp, d_out);
}
